// CrissCrossContrastAttention_9363028705810
// MI455X (gfx1250) — matrix operand order repaired and checked by reading, not compiled, not run on hardware
//
// CrissCrossContrastAttention for MI455X (gfx1250), wave32 + WMMA bf16.
// All GEMM-like convs run on v_wmma_f32_16x16x32_bf16 with f32 accumulators.
// Round 2: A-fragments staged through LDS (stage-1), 2 B-tiles per A-set
// everywhere (8 accumulator fragments per wave) to keep the matrix pipe fed.
#include <hip/hip_runtime.h>
#include <cstdint>

#define B_  16
#define C_  256
#define IC_ 64
#define H_  128
#define W_  128
#define HW_ (H_*W_)

typedef __attribute__((ext_vector_type(16))) __bf16 v16bf;
typedef __attribute__((ext_vector_type(8)))  float  v8f;

static __device__ __forceinline__ unsigned short f2bf(float f) {
  unsigned int u = __float_as_uint(f);
  unsigned int r = u + 0x7FFFu + ((u >> 16) & 1u);   // round-to-nearest-even
  return (unsigned short)(r >> 16);
}
static __device__ __forceinline__ float bf2f(unsigned short h) {
  return __uint_as_float(((unsigned int)h) << 16);
}

#define WMMA_BF16(A, B, Cacc) \
  __builtin_amdgcn_wmma_f32_16x16x32_bf16(false, (A), false, (B), (short)0, (Cacc), false, false)

// ---------------------------------------------------------------- BN stats ---
__global__ void k_bn_stats(const float* __restrict__ x,
                           float* __restrict__ psum, float* __restrict__ psumsq) {
  int plane = blockIdx.x;               // b*256 + c
  int b = plane >> 8, c = plane & 255;
  const float* p = x + ((size_t)(b * C_) + c) * HW_;
  float s = 0.f, s2 = 0.f;
  for (int i = threadIdx.x; i < HW_; i += 256) { float v = p[i]; s += v; s2 += v * v; }
  __shared__ float sh0[256], sh1[256];
  sh0[threadIdx.x] = s; sh1[threadIdx.x] = s2;
  __syncthreads();
  for (int off = 128; off > 0; off >>= 1) {
    if (threadIdx.x < off) { sh0[threadIdx.x] += sh0[threadIdx.x + off];
                             sh1[threadIdx.x] += sh1[threadIdx.x + off]; }
    __syncthreads();
  }
  if (threadIdx.x == 0) { psum[c * B_ + b] = sh0[0]; psumsq[c * B_ + b] = sh1[0]; }
}

__global__ void k_bn_final(const float* __restrict__ psum, const float* __restrict__ psumsq,
                           const float* __restrict__ bnw, const float* __restrict__ bnb,
                           float* __restrict__ scale, float* __restrict__ bias) {
  int c = threadIdx.x;
  float s = 0.f, s2 = 0.f;
  for (int b = 0; b < B_; b++) { s += psum[c * B_ + b]; s2 += psumsq[c * B_ + b]; }
  float inv_n = 1.0f / (float)(B_ * HW_);
  float mean = s * inv_n;
  float var  = s2 * inv_n - mean * mean;
  float rstd = rsqrtf(var + 1e-5f);
  float sc   = bnw[c] * rstd;
  scale[c] = sc;
  bias[c]  = bnb[c] - mean * sc;
}

// --------------------------------------------------- weight fragment packing --
// 16-bit A-matrix (16x32) lane layout (contiguous k): lane = m + 16*(k>>4),
// half-slot = k&15.  Fragments stored as [..frag..][lane][16 halfs].
__global__ void k_pack_s1(const float* __restrict__ w, unsigned short* __restrict__ dst) {
  int i = blockIdx.x * 256 + threadIdx.x;        // 64*256*7 = 114688
  int t = i % 7; int c = (i / 7) & 255; int o = i / (7 * 256);
  int chunk = c >> 5, k = c & 31, mtile = o >> 4, m = o & 15;
  int lane = m + ((k >> 4) << 4);
  int hi   = k & 15;
  dst[((((chunk * 7 + t) * 4 + mtile) * 32 + lane) << 4) + hi] = f2bf(w[i]);
}
// stage-2 weight = transpose(1,0,2,3): WT[co,ic,t] = w[ic,co,t]; M=256, K=64*7
__global__ void k_pack_s2(const float* __restrict__ w, unsigned short* __restrict__ dst) {
  int i = blockIdx.x * 256 + threadIdx.x;        // 256*64*7 = 114688
  int t = i % 7; int ic = (i / 7) & 63; int co = i / (7 * 64);
  int chunk = ic >> 5, k = ic & 31, mtile = co >> 4, m = co & 15;
  int lane = m + ((k >> 4) << 4);
  int hi   = k & 15;
  dst[((((chunk * 7 + t) * 16 + mtile) * 32 + lane) << 4) + hi] =
      f2bf(w[(ic * 256 + co) * 7 + t]);
}

// --------------------------------------------------------- contrast branch ---
__global__ void k_contrast(const float* __restrict__ x, const float* __restrict__ w1,
                           const float* __restrict__ w2, float* __restrict__ cmap) {
  int b = blockIdx.z;
  int h0 = blockIdx.y * 16, w0 = blockIdx.x * 16;
  int tid = threadIdx.x;
  int tr = tid >> 4, tc = tid & 15;
  __shared__ float tile[20][20];
  __shared__ float rs3[20][16], rs5[20][16];
  float acc[64];
#pragma unroll
  for (int o = 0; o < 64; o++) acc[o] = 0.f;
  for (int c = 0; c < C_; c++) {
    const float* xp = x + ((size_t)(b * C_ + c)) * HW_;
    for (int s = tid; s < 400; s += 256) {
      int rr = s / 20, cc = s % 20;
      int gh = h0 + rr - 2, gw = w0 + cc - 2;
      float v = 0.f;
      if ((unsigned)gh < (unsigned)H_ && (unsigned)gw < (unsigned)W_) v = xp[gh * W_ + gw];
      tile[rr][cc] = v;
    }
    __syncthreads();
    for (int s = tid; s < 320; s += 256) {
      int rr = s >> 4, cc = s & 15;
      rs5[rr][cc] = tile[rr][cc] + tile[rr][cc+1] + tile[rr][cc+2] + tile[rr][cc+3] + tile[rr][cc+4];
      rs3[rr][cc] = tile[rr][cc+1] + tile[rr][cc+2] + tile[rr][cc+3];
    }
    __syncthreads();
    float p5 = (rs5[tr][tc] + rs5[tr+1][tc] + rs5[tr+2][tc] + rs5[tr+3][tc] + rs5[tr+4][tc]) * (1.f/25.f);
    float p3 = (rs3[tr+1][tc] + rs3[tr+2][tc] + rs3[tr+3][tc]) * (1.f/9.f);
    float xc = tile[tr+2][tc+2];
    float a3 = fabsf(xc - p3), a5 = fabsf(xc - p5);
#pragma unroll
    for (int o = 0; o < 64; o++)
      acc[o] += w1[o * 512 + c] * a3 + w1[o * 512 + 256 + c] * a5;
    __syncthreads();
  }
  float ssum = 0.f;
#pragma unroll
  for (int o = 0; o < 64; o++) ssum += w2[o] * fmaxf(acc[o], 0.f);
  cmap[(size_t)b * HW_ + (h0 + tr) * W_ + (w0 + tc)] = 1.f / (1.f + expf(-ssum));
}

// ----------------------------------------------------- stage-1 axial convs ---
// Vertical 7x1: block = 64 out-ch x 16 h x 16 w.  Wave r -> output rows
// 2r, 2r+1 (two B tiles per A set); A fragments staged through LDS so all 8
// waves share one cooperative global fetch per chunk.
__global__ void k_s1_v(const float* __restrict__ x, const unsigned short* __restrict__ Af,
                       const float* __restrict__ scale, const float* __restrict__ bias,
                       unsigned short* __restrict__ y) {
  int b = blockIdx.z;
  int h0 = blockIdx.y * 16, w0 = blockIdx.x * 16;
  int tid = threadIdx.x, wave = tid >> 5, lane = tid & 31;
  __shared__ __align__(32) unsigned short ldsB[22][32][16];   // 22.5 KB
  __shared__ __align__(32) unsigned short ldsA[7 * 4 * 32 * 16];  // 28 KB
  v8f acc0a = {}, acc1a = {}, acc2a = {}, acc3a = {};
  v8f acc0b = {}, acc1b = {}, acc2b = {}, acc3b = {};
  const float* xb = x + (size_t)b * C_ * HW_;
  for (int chunk = 0; chunk < 8; chunk++) {
    __syncthreads();
    int c0 = chunk << 5;
    // stage B: 32c x 22 rows x 16 w
    for (int it = 0; it < 44; it++) {
      int s  = tid + (it << 8);
      int n  = s & 15;
      int ri = (s >> 4) % 22;
      int k  = (s >> 4) / 22;
      int c  = c0 + k;
      int hh = h0 + ri - 3;
      float v = 0.f;
      if ((unsigned)hh < (unsigned)H_)
        v = xb[(size_t)c * HW_ + hh * W_ + w0 + n] * scale[c] + bias[c];
      ldsB[ri][n + ((k >> 4) << 4)][k & 15] = f2bf(v);
    }
    // stage A: contiguous copy of this chunk's fragment block (14336 halfs)
    {
      const uint2* src = (const uint2*)(Af + (size_t)chunk * 14336);
      uint2* dstp = (uint2*)ldsA;
      for (int it = 0; it < 7; it++) dstp[tid + (it << 8)] = src[tid + (it << 8)];
    }
    __syncthreads();
#pragma unroll
    for (int t = 0; t < 7; t++) {
      v16bf bm0 = *(const v16bf*)&ldsB[2 * wave + t][lane][0];
      v16bf bm1 = *(const v16bf*)&ldsB[2 * wave + 1 + t][lane][0];
      const unsigned short* ap = ldsA + ((t * 4) * 32 + lane) * 16;
      v16bf a0 = *(const v16bf*)(ap);
      v16bf a1 = *(const v16bf*)(ap + 512);
      v16bf a2 = *(const v16bf*)(ap + 1024);
      v16bf a3 = *(const v16bf*)(ap + 1536);
      acc0a = WMMA_BF16(a0, bm0, acc0a);  acc0b = WMMA_BF16(a0, bm1, acc0b);
      acc1a = WMMA_BF16(a1, bm0, acc1a);  acc1b = WMMA_BF16(a1, bm1, acc1b);
      acc2a = WMMA_BF16(a2, bm0, acc2a);  acc2b = WMMA_BF16(a2, bm1, acc2b);
      acc3a = WMMA_BF16(a3, bm0, acc3a);  acc3b = WMMA_BF16(a3, bm1, acc3b);
    }
  }
  int ha = h0 + 2 * wave, hb = ha + 1;
  int wcol = w0 + (lane & 15), mhi = (lane >> 4) << 3;
  unsigned short* yb = y + (size_t)b * IC_ * HW_;
#pragma unroll
  for (int rr = 0; rr < 8; rr++) {
    int m = rr + mhi;
    yb[(size_t)( 0 + m) * HW_ + ha * W_ + wcol] = f2bf(acc0a[rr]);
    yb[(size_t)(16 + m) * HW_ + ha * W_ + wcol] = f2bf(acc1a[rr]);
    yb[(size_t)(32 + m) * HW_ + ha * W_ + wcol] = f2bf(acc2a[rr]);
    yb[(size_t)(48 + m) * HW_ + ha * W_ + wcol] = f2bf(acc3a[rr]);
    yb[(size_t)( 0 + m) * HW_ + hb * W_ + wcol] = f2bf(acc0b[rr]);
    yb[(size_t)(16 + m) * HW_ + hb * W_ + wcol] = f2bf(acc1b[rr]);
    yb[(size_t)(32 + m) * HW_ + hb * W_ + wcol] = f2bf(acc2b[rr]);
    yb[(size_t)(48 + m) * HW_ + hb * W_ + wcol] = f2bf(acc3b[rr]);
  }
}

// Horizontal 1x7: h/w swapped; wave r -> output columns 2r, 2r+1; N = 16 h.
__global__ void k_s1_h(const float* __restrict__ x, const unsigned short* __restrict__ Af,
                       const float* __restrict__ scale, const float* __restrict__ bias,
                       unsigned short* __restrict__ y) {
  int b = blockIdx.z;
  int w0 = blockIdx.x * 16, h0 = blockIdx.y * 16;
  int tid = threadIdx.x, wave = tid >> 5, lane = tid & 31;
  __shared__ __align__(32) unsigned short ldsB[22][32][16];
  __shared__ __align__(32) unsigned short ldsA[7 * 4 * 32 * 16];
  v8f acc0a = {}, acc1a = {}, acc2a = {}, acc3a = {};
  v8f acc0b = {}, acc1b = {}, acc2b = {}, acc3b = {};
  const float* xb = x + (size_t)b * C_ * HW_;
  for (int chunk = 0; chunk < 8; chunk++) {
    __syncthreads();
    int c0 = chunk << 5;
    for (int it = 0; it < 44; it++) {
      int s  = tid + (it << 8);
      int ci = s % 22;                 // w-column (fastest for coalescing)
      int n  = (s / 22) & 15;          // h index (fragment column)
      int k  = s / 352;                // channel in chunk
      int c  = c0 + k;
      int ww = w0 + ci - 3;
      float v = 0.f;
      if ((unsigned)ww < (unsigned)W_)
        v = xb[(size_t)c * HW_ + (h0 + n) * W_ + ww] * scale[c] + bias[c];
      ldsB[ci][n + ((k >> 4) << 4)][k & 15] = f2bf(v);
    }
    {
      const uint2* src = (const uint2*)(Af + (size_t)chunk * 14336);
      uint2* dstp = (uint2*)ldsA;
      for (int it = 0; it < 7; it++) dstp[tid + (it << 8)] = src[tid + (it << 8)];
    }
    __syncthreads();
#pragma unroll
    for (int t = 0; t < 7; t++) {
      v16bf bm0 = *(const v16bf*)&ldsB[2 * wave + t][lane][0];
      v16bf bm1 = *(const v16bf*)&ldsB[2 * wave + 1 + t][lane][0];
      const unsigned short* ap = ldsA + ((t * 4) * 32 + lane) * 16;
      v16bf a0 = *(const v16bf*)(ap);
      v16bf a1 = *(const v16bf*)(ap + 512);
      v16bf a2 = *(const v16bf*)(ap + 1024);
      v16bf a3 = *(const v16bf*)(ap + 1536);
      acc0a = WMMA_BF16(a0, bm0, acc0a);  acc0b = WMMA_BF16(a0, bm1, acc0b);
      acc1a = WMMA_BF16(a1, bm0, acc1a);  acc1b = WMMA_BF16(a1, bm1, acc1b);
      acc2a = WMMA_BF16(a2, bm0, acc2a);  acc2b = WMMA_BF16(a2, bm1, acc2b);
      acc3a = WMMA_BF16(a3, bm0, acc3a);  acc3b = WMMA_BF16(a3, bm1, acc3b);
    }
  }
  int wa = w0 + 2 * wave, wb = wa + 1;
  int h = h0 + (lane & 15), mhi = (lane >> 4) << 3;
  unsigned short* yb = y + (size_t)b * IC_ * HW_;
#pragma unroll
  for (int rr = 0; rr < 8; rr++) {
    int m = rr + mhi;
    yb[(size_t)( 0 + m) * HW_ + h * W_ + wa] = f2bf(acc0a[rr]);
    yb[(size_t)(16 + m) * HW_ + h * W_ + wa] = f2bf(acc1a[rr]);
    yb[(size_t)(32 + m) * HW_ + h * W_ + wa] = f2bf(acc2a[rr]);
    yb[(size_t)(48 + m) * HW_ + h * W_ + wa] = f2bf(acc3a[rr]);
    yb[(size_t)( 0 + m) * HW_ + h * W_ + wb] = f2bf(acc0b[rr]);
    yb[(size_t)(16 + m) * HW_ + h * W_ + wb] = f2bf(acc1b[rr]);
    yb[(size_t)(32 + m) * HW_ + h * W_ + wb] = f2bf(acc2b[rr]);
    yb[(size_t)(48 + m) * HW_ + h * W_ + wb] = f2bf(acc3b[rr]);
  }
}

// ------------------------------------------------ softmax stats + act_dn -----
__global__ void k_smax(const unsigned short* __restrict__ yv,
                       const unsigned short* __restrict__ yh, float* __restrict__ stats) {
  int id = blockIdx.x;                         // dir*1024 + b*64 + ic
  int dir = id >> 10, plane = id & 1023;
  const unsigned short* p = (dir ? yh : yv) + (size_t)plane * HW_;
  __shared__ float sh[256];
  float m = -3.4e38f;
  for (int i = threadIdx.x; i < HW_; i += 256) m = fmaxf(m, bf2f(p[i]));
  sh[threadIdx.x] = m; __syncthreads();
  for (int off = 128; off > 0; off >>= 1) {
    if (threadIdx.x < off) sh[threadIdx.x] = fmaxf(sh[threadIdx.x], sh[threadIdx.x + off]);
    __syncthreads();
  }
  float mx = sh[0];
  __syncthreads();
  float s = 0.f;
  for (int i = threadIdx.x; i < HW_; i += 256) s += expf(bf2f(p[i]) - mx);
  sh[threadIdx.x] = s; __syncthreads();
  for (int off = 128; off > 0; off >>= 1) {
    if (threadIdx.x < off) sh[threadIdx.x] += sh[threadIdx.x + off];
    __syncthreads();
  }
  if (threadIdx.x == 0) { stats[id * 2] = mx; stats[id * 2 + 1] = sh[0]; }
}

__global__ void k_actdn(unsigned short* __restrict__ yv, unsigned short* __restrict__ yh,
                        const float* __restrict__ stats) {
  long idx = (long)blockIdx.x * 256 + threadIdx.x;  // 2*16*8*16384
  int pix  = (int)(idx & 16383);
  int rest = (int)(idx >> 14);
  int head = rest & 7;
  int b    = (rest >> 3) & 15;
  int dir  = rest >> 7;
  unsigned short* y = dir ? yh : yv;
  float e[8]; float T = 0.f;
#pragma unroll
  for (int j = 0; j < 8; j++) {
    int ic = head * 8 + j;
    int sid = dir * 1024 + b * 64 + ic;
    float mx = stats[sid * 2], sm = stats[sid * 2 + 1];
    e[j] = expf(bf2f(y[((size_t)(b * 64 + ic)) * HW_ + pix]) - mx) / sm;
    T += e[j];
  }
  float inv = 1.f / (T + 1e-6f);
#pragma unroll
  for (int j = 0; j < 8; j++) {
    int ic = head * 8 + j;
    y[((size_t)(b * 64 + ic)) * HW_ + pix] = f2bf(e[j] * inv);
  }
}

// ----------------------------------------------------- stage-2 axial convs ---
// Vertical: M=256 (wave owns a 64-ch quarter = 4 frags) x 2 h-rows; block
// covers 4 h x 16 w.  Each A-quad feeds 8 WMMAs.  Writes raw context_v.
__global__ void k_s2_v(const unsigned short* __restrict__ z, const unsigned short* __restrict__ Af,
                       float* __restrict__ out) {
  int b = blockIdx.z;
  int h0 = blockIdx.y * 4, w0 = blockIdx.x * 16;
  int tid = threadIdx.x, wave = tid >> 5, lane = tid & 31;
  int hp = wave & 1, mq = wave >> 1;
  __shared__ __align__(32) unsigned short ldsB[10][2][32][16];   // 20 KB
  const unsigned short* zb = z + (size_t)b * IC_ * HW_;
  for (int it = 0; it < 40; it++) {              // 64ic x 10 rows x 16 w
    int s  = tid + (it << 8);
    int n  = s & 15;
    int ri = (s >> 4) % 10;
    int ic = (s >> 4) / 10;
    int hh = h0 + ri - 3;
    unsigned short v = 0;
    if ((unsigned)hh < (unsigned)H_) v = zb[(size_t)ic * HW_ + hh * W_ + w0 + n];
    ldsB[ri][ic >> 5][n + (((ic >> 4) & 1) << 4)][ic & 15] = v;
  }
  __syncthreads();
  v8f acc0a = {}, acc1a = {}, acc2a = {}, acc3a = {};
  v8f acc0b = {}, acc1b = {}, acc2b = {}, acc3b = {};
#pragma unroll
  for (int chunk = 0; chunk < 2; chunk++) {
#pragma unroll
    for (int t = 0; t < 7; t++) {
      v16bf bm0 = *(const v16bf*)&ldsB[2 * hp + t][chunk][lane][0];
      v16bf bm1 = *(const v16bf*)&ldsB[2 * hp + 1 + t][chunk][lane][0];
      const unsigned short* ap = Af + ((((size_t)chunk * 7 + t) * 16 + mq * 4) * 32 + lane) * 16;
      v16bf a0 = *(const v16bf*)(ap);
      v16bf a1 = *(const v16bf*)(ap + 512);
      v16bf a2 = *(const v16bf*)(ap + 1024);
      v16bf a3 = *(const v16bf*)(ap + 1536);
      acc0a = WMMA_BF16(a0, bm0, acc0a);  acc0b = WMMA_BF16(a0, bm1, acc0b);
      acc1a = WMMA_BF16(a1, bm0, acc1a);  acc1b = WMMA_BF16(a1, bm1, acc1b);
      acc2a = WMMA_BF16(a2, bm0, acc2a);  acc2b = WMMA_BF16(a2, bm1, acc2b);
      acc3a = WMMA_BF16(a3, bm0, acc3a);  acc3b = WMMA_BF16(a3, bm1, acc3b);
    }
  }
  int ha = h0 + 2 * hp, hb = ha + 1;
  int wcol = w0 + (lane & 15), mhi = (lane >> 4) << 3;
  float* ob = out + (size_t)b * C_ * HW_;
#pragma unroll
  for (int rr = 0; rr < 8; rr++) {
    int m = rr + mhi;
    ob[(size_t)((mq * 4 + 0) * 16 + m) * HW_ + ha * W_ + wcol] = acc0a[rr];
    ob[(size_t)((mq * 4 + 1) * 16 + m) * HW_ + ha * W_ + wcol] = acc1a[rr];
    ob[(size_t)((mq * 4 + 2) * 16 + m) * HW_ + ha * W_ + wcol] = acc2a[rr];
    ob[(size_t)((mq * 4 + 3) * 16 + m) * HW_ + ha * W_ + wcol] = acc3a[rr];
    ob[(size_t)((mq * 4 + 0) * 16 + m) * HW_ + hb * W_ + wcol] = acc0b[rr];
    ob[(size_t)((mq * 4 + 1) * 16 + m) * HW_ + hb * W_ + wcol] = acc1b[rr];
    ob[(size_t)((mq * 4 + 2) * 16 + m) * HW_ + hb * W_ + wcol] = acc2b[rr];
    ob[(size_t)((mq * 4 + 3) * 16 + m) * HW_ + hb * W_ + wcol] = acc3b[rr];
  }
}

// Horizontal stage-2 + fused epilogue: out = x + gamma*(ctx_v+ctx_h)*cmap.
// Wave owns a 64-ch quarter x 2 w-columns; block covers 4 w x 16 h.
__global__ void k_s2_h(const unsigned short* __restrict__ z, const unsigned short* __restrict__ Af,
                       const float* __restrict__ x, const float* __restrict__ cmap,
                       const float* __restrict__ gamma, float* __restrict__ out) {
  int b = blockIdx.z;
  int w0 = blockIdx.x * 4, h0 = blockIdx.y * 16;
  int tid = threadIdx.x, wave = tid >> 5, lane = tid & 31;
  int wp = wave & 1, mq = wave >> 1;
  __shared__ __align__(32) unsigned short ldsB[10][2][32][16];
  const unsigned short* zb = z + (size_t)b * IC_ * HW_;
  for (int it = 0; it < 40; it++) {              // 64ic x 10 cols x 16 h
    int s  = tid + (it << 8);
    int ci = s % 10;
    int n  = (s / 10) & 15;
    int ic = s / 160;
    int ww = w0 + ci - 3;
    unsigned short v = 0;
    if ((unsigned)ww < (unsigned)W_) v = zb[(size_t)ic * HW_ + (h0 + n) * W_ + ww];
    ldsB[ci][ic >> 5][n + (((ic >> 4) & 1) << 4)][ic & 15] = v;
  }
  __syncthreads();
  v8f acc0a = {}, acc1a = {}, acc2a = {}, acc3a = {};
  v8f acc0b = {}, acc1b = {}, acc2b = {}, acc3b = {};
#pragma unroll
  for (int chunk = 0; chunk < 2; chunk++) {
#pragma unroll
    for (int t = 0; t < 7; t++) {
      v16bf bm0 = *(const v16bf*)&ldsB[2 * wp + t][chunk][lane][0];
      v16bf bm1 = *(const v16bf*)&ldsB[2 * wp + 1 + t][chunk][lane][0];
      const unsigned short* ap = Af + ((((size_t)chunk * 7 + t) * 16 + mq * 4) * 32 + lane) * 16;
      v16bf a0 = *(const v16bf*)(ap);
      v16bf a1 = *(const v16bf*)(ap + 512);
      v16bf a2 = *(const v16bf*)(ap + 1024);
      v16bf a3 = *(const v16bf*)(ap + 1536);
      acc0a = WMMA_BF16(a0, bm0, acc0a);  acc0b = WMMA_BF16(a0, bm1, acc0b);
      acc1a = WMMA_BF16(a1, bm0, acc1a);  acc1b = WMMA_BF16(a1, bm1, acc1b);
      acc2a = WMMA_BF16(a2, bm0, acc2a);  acc2b = WMMA_BF16(a2, bm1, acc2b);
      acc3a = WMMA_BF16(a3, bm0, acc3a);  acc3b = WMMA_BF16(a3, bm1, acc3b);
    }
  }
  int wa = w0 + 2 * wp, wb = wa + 1;
  int h = h0 + (lane & 15), mhi = (lane >> 4) << 3;
  float g   = gamma[0];
  float cma = cmap[(size_t)b * HW_ + h * W_ + wa];
  float cmb = cmap[(size_t)b * HW_ + h * W_ + wb];
  const float* xb = x + (size_t)b * C_ * HW_;
  float* ob = out + (size_t)b * C_ * HW_;
#pragma unroll
  for (int rr = 0; rr < 8; rr++) {
    int m = rr + mhi;
#pragma unroll
    for (int f = 0; f < 4; f++) {
      float aa = (f == 0) ? acc0a[rr] : (f == 1) ? acc1a[rr] : (f == 2) ? acc2a[rr] : acc3a[rr];
      float ab = (f == 0) ? acc0b[rr] : (f == 1) ? acc1b[rr] : (f == 2) ? acc2b[rr] : acc3b[rr];
      size_t row = (size_t)((mq * 4 + f) * 16 + m) * HW_ + h * W_;
      float ctxa = aa + ob[row + wa];
      float ctxb = ab + ob[row + wb];
      ob[row + wa] = xb[row + wa] + g * ctxa * cma;
      ob[row + wb] = xb[row + wb] + g * ctxb * cmb;
    }
  }
}

// ------------------------------------------------------------------- launch --
extern "C" void kernel_launch(void* const* d_in, const int* in_sizes, int n_in,
                              void* d_out, int out_size, void* d_ws, size_t ws_size,
                              hipStream_t stream) {
  (void)in_sizes; (void)n_in; (void)out_size; (void)ws_size;
  const float* x     = (const float*)d_in[0];
  const float* w1    = (const float*)d_in[1];
  const float* w2    = (const float*)d_in[2];
  const float* kv    = (const float*)d_in[3];
  const float* kh    = (const float*)d_in[4];
  const float* gamma = (const float*)d_in[5];
  const float* bnw   = (const float*)d_in[6];
  const float* bnb   = (const float*)d_in[7];
  char* ws = (char*)d_ws;
  // workspace layout (bytes)
  float* psum   = (float*)(ws + 0);                 // 4096 f32
  float* psumsq = (float*)(ws + 16384);             // 4096 f32
  float* scale  = (float*)(ws + 32768);             // 256 f32
  float* bias   = (float*)(ws + 33792);             // 256 f32
  float* smax   = (float*)(ws + 34816);             // 2048*2 f32
  float* cmap   = (float*)(ws + 51200);             // 16*16384 f32 = 1 MB
  unsigned short* wvf  = (unsigned short*)(ws + 1099776);            // 224 KB each
  unsigned short* whf  = (unsigned short*)(ws + 1099776 + 229376);
  unsigned short* wvtf = (unsigned short*)(ws + 1099776 + 2 * 229376);
  unsigned short* whtf = (unsigned short*)(ws + 1099776 + 3 * 229376);
  unsigned short* yv   = (unsigned short*)(ws + 2097152);            // 32 MB
  unsigned short* yh   = (unsigned short*)(ws + 2097152 + 33554432); // 32 MB
  float* out = (float*)d_out;

  k_bn_stats<<<4096, 256, 0, stream>>>(x, psum, psumsq);
  k_bn_final<<<1, 256, 0, stream>>>(psum, psumsq, bnw, bnb, scale, bias);
  k_pack_s1<<<448, 256, 0, stream>>>(kv, wvf);
  k_pack_s1<<<448, 256, 0, stream>>>(kh, whf);
  k_pack_s2<<<448, 256, 0, stream>>>(kv, wvtf);
  k_pack_s2<<<448, 256, 0, stream>>>(kh, whtf);
  k_contrast<<<dim3(8, 8, 16), 256, 0, stream>>>(x, w1, w2, cmap);
  k_s1_v<<<dim3(8, 8, 16), 256, 0, stream>>>(x, wvf, scale, bias, yv);
  k_s1_h<<<dim3(8, 8, 16), 256, 0, stream>>>(x, whf, scale, bias, yh);
  k_smax<<<2048, 256, 0, stream>>>(yv, yh, smax);
  k_actdn<<<16384, 256, 0, stream>>>(yv, yh, smax);
  k_s2_v<<<dim3(8, 32, 16), 256, 0, stream>>>(yv, wvtf, out);
  k_s2_h<<<dim3(32, 8, 16), 256, 0, stream>>>(yh, whtf, x, cmap, gamma, out);
}
